// FreeConvNetwork_38044820308051
// MI455X (gfx1250) — matrix operand order repaired and checked by reading, not compiled, not run on hardware
//
#include <hip/hip_runtime.h>
#include <hip/hip_bf16.h>

typedef __attribute__((ext_vector_type(16))) __bf16 v16bf;
typedef __attribute__((ext_vector_type(8)))  float  v8f;

// ---------------------------------------------------------------------------
// WMMA 16x16x32 bf16 fragment loaders (layouts per CDNA5 ISA 7.12.2, wave32)
// A: 16(M) x 32(K), row-major in LDS, row stride `ld` elements.
//   lane 0-15:  M=lane,    K base 0  (VGPR0-3: K 0-7,  VGPR4-7: K 16-23)
//   lane 16-31: M=lane-16, K base 8  (VGPR0-3: K 8-15, VGPR4-7: K 24-31)
// ---------------------------------------------------------------------------
__device__ __forceinline__ v16bf load_a_frag(const __bf16* A, int ld) {
  const int lane = threadIdx.x & 31;
  const int m  = lane & 15;
  const int kb = (lane >> 4) << 3;          // 0 or 8
  const __bf16* row = A + m * ld;
  v16bf a;
#pragma unroll
  for (int i = 0; i < 16; ++i) {
    int k = kb + i + ((i >= 8) ? 8 : 0);    // 0-7 -> kb+0..7 ; 8-15 -> kb+16..23
    a[i] = row[k];
  }
  return a;
}

// B: 32(K) x 16(N); stored transposed in LDS as Bs[n][k] (row stride ld).
//   lane 0-15:  N=lane,    K 0-7,16-23 ; lane 16-31: N=lane-16, K 8-15,24-31
__device__ __forceinline__ v16bf load_b_frag(const __bf16* B, int ld) {
  const int lane = threadIdx.x & 31;
  const int n  = lane & 15;
  const int kb = (lane >> 4) << 3;          // 0 or 8
  const __bf16* row = B + n * ld;
  v16bf b;
#pragma unroll
  for (int i = 0; i < 16; ++i) b[i] = row[kb + i + ((i >= 8) ? 8 : 0)];
  return b;
}

// ---------------------------------------------------------------------------
// Locally-connected conv layer as per-location WMMA GEMM.
//   out[b,o,i,j] = relu( (sum_{c,p} w[o,c,i,j,p]*patch[b,c,i,j,p] + lcb[o,i,j])
//                        * bns[o] + bnt[o] )
// grid = OH*OW*MSPLIT blocks of 256 threads (8 waves). Each block handles
// OB = O/MSPLIT output channels x 64 batch at one location; K = C*KSZ^2
// reduced in 32-wide LDS slabs (fp32 -> bf16 in-register; bf16 keeps the
// fp32 exponent). Staging processes 2 adjacent K elements per thread so the
// compiler emits packed v_cvt_pk_bf16_f32 + merged ds_store_b32.
// ---------------------------------------------------------------------------
template<int O, int C, int KSZ, int STRIDE, int IH, int IW, int OH, int OW,
         bool BN0, int MSPLIT>
__launch_bounds__(256)
__global__ void lc_kernel(const float* __restrict__ in,   // [64,C,IH,IW]
                          const float* __restrict__ w,    // [O,C,OH,OW,KSZ*KSZ]
                          const float* __restrict__ lcb,  // [O,OH,OW]
                          const float* __restrict__ bns,  // [O]
                          const float* __restrict__ bnt,  // [O]
                          const float* __restrict__ s0,   // [C] (BN0 only)
                          const float* __restrict__ t0,   // [C] (BN0 only)
                          float* __restrict__ out) {      // [64,O,OH,OW]
  constexpr int K2 = KSZ * KSZ;
  constexpr int K  = C * K2;
  constexpr int KP = (K + 31) & ~31;
  constexpr bool PAD = (KP != K);           // only lc1 (K=300)
  constexpr int OB = O / MSPLIT;            // rows this block owns
  constexpr int NT = OB / 32;               // 16x16 tiles per wave

  __shared__ __bf16 As[OB * 32];            // A slab: OB x 32 (row-major)
  __shared__ __bf16 Bs[64 * 32];            // B slab: 64(b) x 32(k)

  const int tid  = threadIdx.x;
  const int wave = tid >> 5;
  const int loc  = blockIdx.x / MSPLIT;
  const int o_base = (blockIdx.x % MSPLIT) * OB;
  const int i = loc / OW, j = loc % OW;

  v8f acc[NT] = {};

  for (int kk = 0; kk < KP; kk += 32) {
    // ---- stage weights: As[ol][kc] = w[o_base+ol, c(kidx), i, j, p(kidx)]
    for (int e = tid; e < OB * 16; e += 256) {
      const int ol = e >> 4;
      const int kc = (e & 15) << 1;         // invariant per thread -> div hoisted
      const int o  = o_base + ol;
      float v[2];
#pragma unroll
      for (int u = 0; u < 2; ++u) {
        const int kidx = kk + kc + u;
        const int kq = (!PAD || kidx < K) ? kidx : (K - 1);   // branchless clamp
        const int c = kq / K2, p = kq % K2;
        const float t = w[(((o * C + c) * OH + i) * OW + j) * K2 + p];
        v[u] = (!PAD || kidx < K) ? t : 0.f;
      }
      As[(ol << 5) + kc]     = (__bf16)v[0];
      As[(ol << 5) + kc + 1] = (__bf16)v[1];
    }
    // ---- stage patches: Bs[b][kc] = in[b, c, i*S+di, j*S+dj] (bn0 folded)
    for (int e = tid; e < 64 * 16; e += 256) {
      const int b  = e >> 4;
      const int kc = (e & 15) << 1;
      float v[2];
#pragma unroll
      for (int u = 0; u < 2; ++u) {
        const int kidx = kk + kc + u;
        const int kq = (!PAD || kidx < K) ? kidx : (K - 1);
        const int c = kq / K2, p = kq % K2;
        const int di = p / KSZ, dj = p % KSZ;
        float t = in[((b * C + c) * IH + (i * STRIDE + di)) * IW + (j * STRIDE + dj)];
        if (BN0) t = t * s0[c] + t0[c];
        v[u] = (!PAD || kidx < K) ? t : 0.f;
      }
      Bs[(b << 5) + kc]     = (__bf16)v[0];
      Bs[(b << 5) + kc + 1] = (__bf16)v[1];
    }
    __syncthreads();
#pragma unroll
    for (int r = 0; r < NT; ++r) {
      const int t = wave + 8 * r;
      const int mt = t >> 2, nt = t & 3;
      v16bf a  = load_a_frag(&As[mt * 16 * 32], 32);
      v16bf bb = load_b_frag(&Bs[nt * 16 * 32], 32);
      acc[r] = __builtin_amdgcn_wmma_f32_16x16x32_bf16(
          false, a, false, bb, (short)0, acc[r], false, false);
    }
    __syncthreads();
  }

  // ---- epilogue: +local bias, BN affine, ReLU, scatter to [b,o,i,j]
  const int lane = tid & 31;
  const int n  = lane & 15;
  const int mh = (lane >> 4) << 3;
#pragma unroll
  for (int r = 0; r < NT; ++r) {
    const int t = wave + 8 * r;
    const int mt = t >> 2, nt = t & 3;
    const int b = nt * 16 + n;
#pragma unroll
    for (int e = 0; e < 8; ++e) {
      const int o = o_base + mt * 16 + mh + e;
      float v = acc[r][e] + lcb[(o * OH + i) * OW + j];
      v = fmaxf(v * bns[o] + bnt[o], 0.f);
      out[((b * O + o) * OH + i) * OW + j] = v;
    }
  }
}

// ---------------------------------------------------------------------------
// Fully-connected: out[b,o] = act( sum_k h[b,k]*w[o,k] + bias[o] ).
// 32 output rows per block -> 128 blocks for the 4096-row layers (keeps
// enough WGPs busy to stream fc2's 67 MB at HBM rate). Paired K staging:
// adjacent fp32 loads merge to global_load_b64, packed bf16 convert.
// ---------------------------------------------------------------------------
template<int MTOT, int K, bool RELU>
__launch_bounds__(256)
__global__ void fc_kernel(const float* __restrict__ w,    // [MTOT,K]
                          const float* __restrict__ bias, // [MTOT]
                          const float* __restrict__ h,    // [64,K]
                          float* __restrict__ out) {      // [64,MTOT]
  constexpr int MBLK = 32;                  // 2 M-tiles x 4 N-tiles = 8 waves
  __shared__ __bf16 As[MBLK * 32];
  __shared__ __bf16 Bs[64 * 32];

  const int tid  = threadIdx.x;
  const int wave = tid >> 5;
  const int m0   = blockIdx.x * MBLK;

  v8f acc = {};

  for (int kk = 0; kk < K; kk += 32) {
    for (int e = tid; e < MBLK * 16; e += 256) {
      const int ol = e >> 4;
      const int kc = (e & 15) << 1;
      const int o  = m0 + ol;
      const bool ok = (o < MTOT);
      const int oq = ok ? o : (MTOT - 1);
      const float v0 = w[oq * K + kk + kc];         // adjacent pair -> b64
      const float v1 = w[oq * K + kk + kc + 1];
      As[(ol << 5) + kc]     = (__bf16)(ok ? v0 : 0.f);
      As[(ol << 5) + kc + 1] = (__bf16)(ok ? v1 : 0.f);
    }
    for (int e = tid; e < 64 * 16; e += 256) {
      const int b  = e >> 4;
      const int kc = (e & 15) << 1;
      Bs[(b << 5) + kc]     = (__bf16)h[b * K + kk + kc];
      Bs[(b << 5) + kc + 1] = (__bf16)h[b * K + kk + kc + 1];
    }
    __syncthreads();
    {
      const int mt = wave >> 2, nt = wave & 3;
      v16bf a  = load_a_frag(&As[mt * 16 * 32], 32);
      v16bf bb = load_b_frag(&Bs[nt * 16 * 32], 32);
      acc = __builtin_amdgcn_wmma_f32_16x16x32_bf16(
          false, a, false, bb, (short)0, acc, false, false);
    }
    __syncthreads();
  }

  const int lane = tid & 31;
  const int n  = lane & 15;
  const int mh = (lane >> 4) << 3;
  const int mt = wave >> 2, nt = wave & 3;
  const int b = nt * 16 + n;
#pragma unroll
  for (int e = 0; e < 8; ++e) {
    const int o = m0 + mt * 16 + mh + e;
    if (o < MTOT) {
      float v = acc[e] + bias[o];
      if (RELU) v = fmaxf(v, 0.f);
      out[b * MTOT + o] = v;
    }
  }
}

// ---------------------------------------------------------------------------
// MaxPool2d (win x win, stride, pad; -inf padding == clipped window).
// One thread per output element. (B,C,OH,OW) flatten == memory no-op.
// ---------------------------------------------------------------------------
template<int C, int IH, int IW, int OH, int OW, int WIN, int STRIDE, int PAD>
__global__ void maxpool_kernel(const float* __restrict__ in, float* __restrict__ out) {
  const int idx = blockIdx.x * blockDim.x + threadIdx.x;
  const int total = 64 * C * OH * OW;
  if (idx >= total) return;
  int t = idx;
  const int oj = t % OW; t /= OW;
  const int oi = t % OH; t /= OH;
  const int c  = t % C;
  const int b  = t / C;
  const int i0 = oi * STRIDE - PAD, j0 = oj * STRIDE - PAD;
  float m = -INFINITY;
#pragma unroll
  for (int u = 0; u < WIN; ++u) {
    const int ii = i0 + u;
    if (ii < 0 || ii >= IH) continue;
#pragma unroll
    for (int v = 0; v < WIN; ++v) {
      const int jj = j0 + v;
      if (jj < 0 || jj >= IW) continue;
      m = fmaxf(m, in[((b * C + c) * IH + ii) * IW + jj]);
    }
  }
  out[((b * C + c) * OH + oi) * OW + oj] = m;
}

// ---------------------------------------------------------------------------
extern "C" void kernel_launch(void* const* d_in, const int* in_sizes, int n_in,
                              void* d_out, int out_size, void* d_ws, size_t ws_size,
                              hipStream_t stream) {
  const float* x     = (const float*)d_in[0];
  const float* bn0_s = (const float*)d_in[1];
  const float* bn0_t = (const float*)d_in[2];
  const float* lc1_w = (const float*)d_in[3];
  const float* lc1_b = (const float*)d_in[4];
  const float* bn1_s = (const float*)d_in[5];
  const float* bn1_t = (const float*)d_in[6];
  const float* lc2_w = (const float*)d_in[7];
  const float* lc2_b = (const float*)d_in[8];
  const float* bn2_s = (const float*)d_in[9];
  const float* bn2_t = (const float*)d_in[10];
  const float* lc3_w = (const float*)d_in[11];
  const float* lc3_b = (const float*)d_in[12];
  const float* bn3_s = (const float*)d_in[13];
  const float* bn3_t = (const float*)d_in[14];
  const float* lc4_w = (const float*)d_in[15];
  const float* lc4_b = (const float*)d_in[16];
  const float* bn4_s = (const float*)d_in[17];
  const float* bn4_t = (const float*)d_in[18];
  const float* lc5_w = (const float*)d_in[19];
  const float* lc5_b = (const float*)d_in[20];
  const float* bn5_s = (const float*)d_in[21];
  const float* bn5_t = (const float*)d_in[22];
  const float* fc1_w = (const float*)d_in[23];
  const float* fc1_b = (const float*)d_in[24];
  const float* fc2_w = (const float*)d_in[25];
  const float* fc2_b = (const float*)d_in[26];
  const float* fc3_w = (const float*)d_in[27];
  const float* fc3_b = (const float*)d_in[28];

  // workspace layout (fp32 elements), every buffer fully overwritten
  float* ws = (float*)d_ws;
  float* h1 = ws;                        // 64*32*54*54  = 5,971,968
  float* p1 = h1 + 5971968;              // 64*32*27*27  = 1,492,992
  float* h2 = p1 + 1492992;              // 64*96*23*23  = 3,250,176
  float* p2 = h2 + 3250176;              // 64*96*12*12  =   884,736
  float* h3 = p2 + 884736;               // 64*192*10*10 = 1,228,800
  float* h4 = h3 + 1228800;              // 64*128*8*8   =   524,288
  float* h5 = h4 + 524288;               // 64*128*6*6   =   294,912
  float* p5 = h5 + 294912;               // 64*512       =    32,768
  float* f1 = p5 + 32768;                // 64*4096      =   262,144
  float* f2 = f1 + 262144;               // 64*4096      =   262,144

  // lc1 (bn0 fused into patch gather) -> bn1+relu fused epilogue
  lc_kernel<32, 3, 10, 4, 222, 222, 54, 54, true, 1>
      <<<54 * 54, 256, 0, stream>>>(x, lc1_w, lc1_b, bn1_s, bn1_t, bn0_s, bn0_t, h1);
  maxpool_kernel<32, 54, 54, 27, 27, 3, 2, 1>
      <<<(64 * 32 * 27 * 27 + 255) / 256, 256, 0, stream>>>(h1, p1);

  lc_kernel<96, 32, 5, 1, 27, 27, 23, 23, false, 1>
      <<<23 * 23, 256, 0, stream>>>(p1, lc2_w, lc2_b, bn2_s, bn2_t, nullptr, nullptr, h2);
  maxpool_kernel<96, 23, 23, 12, 12, 3, 2, 1>
      <<<(64 * 96 * 12 * 12 + 255) / 256, 256, 0, stream>>>(h2, p2);

  // small layers: M-split to keep more WGPs streaming weights
  lc_kernel<192, 96, 3, 1, 12, 12, 10, 10, false, 2>
      <<<10 * 10 * 2, 256, 0, stream>>>(p2, lc3_w, lc3_b, bn3_s, bn3_t, nullptr, nullptr, h3);
  lc_kernel<128, 192, 3, 1, 10, 10, 8, 8, false, 2>
      <<<8 * 8 * 2, 256, 0, stream>>>(h3, lc4_w, lc4_b, bn4_s, bn4_t, nullptr, nullptr, h4);
  lc_kernel<128, 128, 3, 1, 8, 8, 6, 6, false, 4>
      <<<6 * 6 * 4, 256, 0, stream>>>(h4, lc5_w, lc5_b, bn5_s, bn5_t, nullptr, nullptr, h5);
  maxpool_kernel<128, 6, 6, 2, 2, 3, 3, 1>
      <<<(64 * 128 * 2 * 2 + 255) / 256, 256, 0, stream>>>(h5, p5);

  fc_kernel<4096, 512, true><<<128, 256, 0, stream>>>(fc1_w, fc1_b, p5, f1);
  fc_kernel<4096, 4096, true><<<128, 256, 0, stream>>>(fc2_w, fc2_b, f1, f2);
  fc_kernel<500, 4096, false><<<16, 256, 0, stream>>>(fc3_w, fc3_b, f2, (float*)d_out);
}
